// CFDKT_61400852463826
// MI455X (gfx1250) — matrix operand order repaired and checked by reading, not compiled, not run on hardware
//
#include <hip/hip_runtime.h>
#include <hip/hip_bf16.h>
#include <math.h>

typedef __attribute__((ext_vector_type(16))) __bf16 v16bf;
typedef __attribute__((ext_vector_type(8)))  __bf16 v8bf;
typedef __attribute__((ext_vector_type(8)))  float  v8f;

#define B_      128
#define T_      200
#define NUMC    1024
#define EMB_    256
#define DIN_    384          // EMB + 128 one-hot
#define H_      256
#define G4_     1024         // 4*H
#define M_TOTAL (B_ * T_)    // 25600

__device__ __forceinline__ float sigm(float x) { return 1.0f / (1.0f + __expf(-x)); }

// ---------------- f32 -> bf16 weight conversion ----------------
__global__ void f32_to_bf16_kernel(const float* __restrict__ in,
                                   __bf16* __restrict__ out, int n) {
    int i = blockIdx.x * blockDim.x + threadIdx.x;
    if (i < n) out[i] = (__bf16)in[i];
}

// ---------------- theta construction (c_integration) ----------------
// mode 0: vt = E[c + NUMC*r]   (embedding gather)
// mode 1: vt = h_all[m]        (LSTM hidden)
// theta[m] = concat( vt * (cembT[g0] + cembT[32+g1] + cembT[64+g2]), one-hots )  as bf16
__global__ void theta_kernel(const float* __restrict__ vt_rows,
                             const int* __restrict__ xidx, const int* __restrict__ ridx,
                             const int* __restrict__ g0, const int* __restrict__ g1,
                             const int* __restrict__ g2,
                             const float* __restrict__ cembW,   // [EMB, 128] row-major
                             __bf16* __restrict__ theta, int mode) {
    const int m   = blockIdx.x;
    const int tid = threadIdx.x;       // 128 threads
    const int ig = g0[m], is_ = g1[m], ip = g2[m];
    const float* vrow;
    if (mode == 0) vrow = vt_rows + (size_t)(xidx[m] + NUMC * ridx[m]) * EMB_;
    else           vrow = vt_rows + (size_t)m * EMB_;
    const size_t base = (size_t)m * DIN_;
    for (int e = tid; e < EMB_; e += 128) {
        float cct = cembW[e * 128 + ig] + cembW[e * 128 + 32 + is_]
                  + cembW[e * 128 + 64 + ip];
        theta[base + e] = (__bf16)(vrow[e] * cct);
    }
    float oh = (tid == ig || tid == 32 + is_ || tid == 64 + ip) ? 1.0f : 0.0f;
    theta[base + EMB_ + tid] = (__bf16)oh;
}

// ---------------- WMMA bf16 GEMM:  out[M,N] = A[M,K] * W[N,K]^T + bias (+sigmoid) ----
// Block: 256 threads = 8 waves; block tile 64(M) x 128(N); wave tile 16 x 64.
// A fragment (16x32 bf16): lane = row M, elem e -> K = kh*16+e
// B fragment (32x16 bf16): lane = col N, elem e -> K = kh*16+e   (kh = lane>>4)
// C/D (16x16 f32): lane = col N, vgpr r -> row r + 8*kh
__global__ __launch_bounds__(256) void wmma_gemm_kernel(
        const __bf16* __restrict__ A, const __bf16* __restrict__ W,
        const float* __restrict__ bias0, const float* __restrict__ bias1,
        float* __restrict__ out, int M, int N, int K, int act) {
    const int tid  = threadIdx.x;
    const int wave = tid >> 5;
    const int lane = tid & 31;
    const int lm   = lane & 15;
    const int kh   = lane >> 4;
    const int mbase = blockIdx.x * 64  + (wave & 3) * 16;
    const int nbase = blockIdx.y * 128 + (wave >> 2) * 64;

    v8f acc[4];
    for (int j = 0; j < 4; ++j)
        for (int r2 = 0; r2 < 8; ++r2) acc[j][r2] = 0.0f;

    const __bf16* arow = A + (size_t)(mbase + lm) * K;
    for (int k0 = 0; k0 < K; k0 += 32) {
        v8bf a0 = *(const v8bf*)(arow + k0 + kh * 16);
        v8bf a1 = *(const v8bf*)(arow + k0 + kh * 16 + 8);
        v16bf av;
        for (int e = 0; e < 8; ++e) { av[e] = a0[e]; av[e + 8] = a1[e]; }
        for (int j = 0; j < 4; ++j) {
            const int n = nbase + j * 16 + lm;
            v16bf bv = *(const v16bf*)(W + (size_t)n * K + k0 + kh * 16);
            acc[j] = __builtin_amdgcn_wmma_f32_16x16x32_bf16(
                         false, av, false, bv, (short)0, acc[j], false, false);
        }
    }
    for (int j = 0; j < 4; ++j) {
        const int n = nbase + j * 16 + lm;
        float bsum = bias0[n] + (bias1 ? bias1[n] : 0.0f);
        for (int r2 = 0; r2 < 8; ++r2) {
            const int mrow = mbase + r2 + 8 * kh;
            float v = acc[j][r2] + bsum;
            if (act) v = sigm(v);
            out[(size_t)mrow * N + n] = v;
        }
    }
}

// ---------------- LSTM scan ----------------
// 8 blocks (16 batch rows each), 512 threads = 16 waves (wave w -> gate cols w*64..w*64+63).
// Per step: gates = xg[t] + h @ W_hh^T via WMMA (h bf16 in LDS), activations, update c (regs),
// write h to LDS (bf16, next step's A operand) and to global (f32, for theta_out).
__global__ __launch_bounds__(512) void lstm_kernel(
        const float* __restrict__ xg,    // [B*T, 1024], m = b*T + t
        const __bf16* __restrict__ Whh,  // [1024, 256] bf16
        float* __restrict__ h_all) {     // [B*T, 256]
    __shared__ alignas(16) __bf16 sh_h[16][264];    // h tile, padded rows
    __shared__ alignas(16) __bf16 sh_g[16][1032];   // pre-activation gates, padded rows

    const int tid  = threadIdx.x;
    const int b0   = blockIdx.x * 16;
    const int wave = tid >> 5;
    const int lane = tid & 31;
    const int lm   = lane & 15;
    const int kh   = lane >> 4;
    const int nbase = wave * 64;

    if (tid < 256)
        for (int ii = 0; ii < 16; ++ii) sh_h[ii][tid] = (__bf16)0.0f;

    const int jcol  = tid & 255;   // hidden index handled by this thread
    const int rhalf = tid >> 8;    // which 8 batch rows
    float c_reg[8];
    for (int ii = 0; ii < 8; ++ii) c_reg[ii] = 0.0f;
    __syncthreads();

    for (int t = 0; t < T_; ++t) {
        // init accumulators with precomputed input projection xg (already biased)
        v8f acc[4];
        for (int j = 0; j < 4; ++j) {
            const int n = nbase + j * 16 + lm;
            for (int r2 = 0; r2 < 8; ++r2) {
                const int mrow = r2 + 8 * kh;
                acc[j][r2] = xg[((size_t)(b0 + mrow) * T_ + t) * G4_ + n];
            }
        }
        // gates += h @ W_hh^T
        for (int k0 = 0; k0 < H_; k0 += 32) {
            v8bf a0 = *(const v8bf*)(&sh_h[lm][k0 + kh * 16]);
            v8bf a1 = *(const v8bf*)(&sh_h[lm][k0 + kh * 16 + 8]);
            v16bf av;
            for (int e = 0; e < 8; ++e) { av[e] = a0[e]; av[e + 8] = a1[e]; }
            for (int j = 0; j < 4; ++j) {
                const int n = nbase + j * 16 + lm;
                v16bf bv = *(const v16bf*)(Whh + (size_t)n * H_ + k0 + kh * 16);
                acc[j] = __builtin_amdgcn_wmma_f32_16x16x32_bf16(
                             false, av, false, bv, (short)0, acc[j], false, false);
            }
        }
        // spill gates to LDS for the cross-wave i/f/g/o recombination
        for (int j = 0; j < 4; ++j) {
            const int n = nbase + j * 16 + lm;
            for (int r2 = 0; r2 < 8; ++r2)
                sh_g[r2 + 8 * kh][n] = (__bf16)acc[j][r2];
        }
        __syncthreads();   // gates visible; all sh_h reads of this step are done

        for (int ii = 0; ii < 8; ++ii) {
            const int row = rhalf * 8 + ii;
            float i_g = (float)sh_g[row][jcol];
            float f_g = (float)sh_g[row][256 + jcol];
            float g_g = (float)sh_g[row][512 + jcol];
            float o_g = (float)sh_g[row][768 + jcol];
            float cc = sigm(f_g) * c_reg[ii] + sigm(i_g) * tanhf(g_g);
            c_reg[ii] = cc;
            float hh = sigm(o_g) * tanhf(cc);
            h_all[((size_t)(b0 + row) * T_ + t) * H_ + jcol] = hh;
            sh_h[row][jcol] = (__bf16)hh;
        }
        __syncthreads();   // h updated before next step's WMMA reads
    }
}

// ---------------- launch ----------------
extern "C" void kernel_launch(void* const* d_in, const int* in_sizes, int n_in,
                              void* d_out, int out_size, void* d_ws, size_t ws_size,
                              hipStream_t stream) {
    (void)in_sizes; (void)n_in; (void)out_size; (void)ws_size;
    const int*   c       = (const int*)d_in[0];
    const int*   r       = (const int*)d_in[1];
    const int*   rgap    = (const int*)d_in[2];
    const int*   sgap    = (const int*)d_in[3];
    const int*   pcount  = (const int*)d_in[4];
    const int*   srgap   = (const int*)d_in[5];
    const int*   ssgap   = (const int*)d_in[6];
    const int*   spcount = (const int*)d_in[7];
    const float* E       = (const float*)d_in[8];
    const float* cembW   = (const float*)d_in[9];
    const float* W_ih    = (const float*)d_in[10];
    const float* W_hh    = (const float*)d_in[11];
    const float* b_ih    = (const float*)d_in[12];
    const float* b_hh    = (const float*)d_in[13];
    const float* out_W   = (const float*)d_in[14];
    const float* out_b   = (const float*)d_in[15];

    char* ws = (char*)d_ws;
    size_t off = 0;
    auto alloc = [&](size_t bytes) -> void* {
        void* p = ws + off;
        off += bytes;
        off = (off + 255) & ~(size_t)255;
        return p;
    };
    __bf16* wWih  = (__bf16*)alloc((size_t)G4_ * DIN_ * 2);
    __bf16* wWhh  = (__bf16*)alloc((size_t)G4_ * H_ * 2);
    __bf16* wOutW = (__bf16*)alloc((size_t)NUMC * DIN_ * 2);
    __bf16* theta = (__bf16*)alloc((size_t)M_TOTAL * DIN_ * 2);
    float*  xg    = (float*)alloc((size_t)M_TOTAL * G4_ * 4);
    float*  h_all = (float*)alloc((size_t)M_TOTAL * H_ * 4);

    // 1) weights -> bf16
    f32_to_bf16_kernel<<<(G4_ * DIN_ + 255) / 256, 256, 0, stream>>>(W_ih, wWih, G4_ * DIN_);
    f32_to_bf16_kernel<<<(G4_ * H_   + 255) / 256, 256, 0, stream>>>(W_hh, wWhh, G4_ * H_);
    f32_to_bf16_kernel<<<(NUMC * DIN_ + 255) / 256, 256, 0, stream>>>(out_W, wOutW, NUMC * DIN_);

    // 2) theta_in
    theta_kernel<<<M_TOTAL, 128, 0, stream>>>(E, c, r, rgap, sgap, pcount, cembW, theta, 0);

    // 3) xg = theta_in @ W_ih^T + (b_ih + b_hh)
    wmma_gemm_kernel<<<dim3(M_TOTAL / 64, G4_ / 128), 256, 0, stream>>>(
        theta, wWih, b_ih, b_hh, xg, M_TOTAL, G4_, DIN_, 0);

    // 4) LSTM recurrence (batch-tiled)
    lstm_kernel<<<B_ / 16, 512, 0, stream>>>(xg, wWhh, h_all);

    // 5) theta_out (reuse theta buffer)
    theta_kernel<<<M_TOTAL, 128, 0, stream>>>(h_all, nullptr, nullptr,
                                              srgap, ssgap, spcount, cembW, theta, 1);

    // 6) y = sigmoid(theta_out @ out_W^T + out_b)
    wmma_gemm_kernel<<<dim3(M_TOTAL / 64, NUMC / 128), 256, 0, stream>>>(
        theta, wOutW, out_b, nullptr, (float*)d_out, M_TOTAL, NUMC, DIN_, 1);
}
